// GuidanceClassifier_64742337020211
// MI455X (gfx1250) — matrix operand order repaired and checked by reading, not compiled, not run on hardware
//
#include <hip/hip_runtime.h>
#include <hip/hip_bf16.h>

// ---------------------------------------------------------------------------
// RGCN forward for MI455X / gfx1250, wave32 + WMMA bf16 + async LDS copy.
//   N=100000 nodes, E=600000 edges, D=128, R=8 relations, B=64 graphs, L=3.
// Per layer: zero(sums,cnt) -> edge scatter (f32 atomics) -> fused GEMM:
//   xout = relu( (sums*1/cnt)[N,1024] @ Wrel[1024,128] + x @ Wroot + bias )
// Weights are pre-converted once to bf16 in transposed per-chunk layout so the
// B tile staging is a pure async global->LDS byte copy (ASYNCcnt path).
// ---------------------------------------------------------------------------

#define D_DIM 128
#define R_REL 8
#define L_LAY 3
#define M_TILE 80          // rows per block (100000 / 80 = 1250 blocks, exact)

typedef __attribute__((ext_vector_type(16))) __bf16 v16bf;
typedef __attribute__((ext_vector_type(8)))  float  v8f;

// ---------------- fill ----------------
__global__ void k_fill_zero(float* __restrict__ p, size_t n) {
    size_t i = (size_t)blockIdx.x * blockDim.x + threadIdx.x;
    size_t stride = (size_t)gridDim.x * blockDim.x;
    for (; i < n; i += stride) p[i] = 0.0f;
}

// ---------------- embedding gather: x[n,d] = emb[node_type[n], d] ----------
__global__ void k_gather_embed(const int* __restrict__ node_type,
                               const float* __restrict__ emb,
                               float* __restrict__ x, int N) {
    int t = blockIdx.x * blockDim.x + threadIdx.x;
    if (t >= N * D_DIM) return;
    int n = t >> 7, d = t & 127;
    x[t] = emb[(size_t)node_type[n] * D_DIM + d];
}

// ---------------- weight prep: f32 [K,128] -> bf16 transposed chunks -------
// Layout: wt[((l*9 + kt)*128 + col)*128 + kp], kt 0..7 = Wrel chunks, 8 = Wroot.
__global__ void k_prep_w(const float* __restrict__ rel_w,
                         const float* __restrict__ root_w,
                         __bf16* __restrict__ wt) {
    const int per_layer = 9 * 128 * 128;
    int t = blockIdx.x * blockDim.x + threadIdx.x;
    if (t >= L_LAY * per_layer) return;
    int l   = t / per_layer;
    int r   = t % per_layer;
    int kt  = r / (128 * 128);
    int r2  = r % (128 * 128);
    int col = r2 / 128;
    int kp  = r2 % 128;
    float v;
    if (kt < 8) v = rel_w[(size_t)l * R_REL * D_DIM * D_DIM + (size_t)(kt * 128 + kp) * 128 + col];
    else        v = root_w[(size_t)l * D_DIM * D_DIM + (size_t)kp * 128 + col];
    wt[t] = (__bf16)v;
}

// ---------------- edge scatter: sums[(dst*R+et), d] += x[src, d] ----------
__global__ void k_scatter(const int* __restrict__ src, const int* __restrict__ dst,
                          const int* __restrict__ etype, const float* __restrict__ x,
                          float* __restrict__ sums, float* __restrict__ cnt, int E) {
    int t = blockIdx.x * blockDim.x + threadIdx.x;
    if (t >= E * D_DIM) return;
    int e = t >> 7, d = t & 127;
    int seg = dst[e] * R_REL + etype[e];
    atomicAdd(&sums[(size_t)seg * D_DIM + d], x[(size_t)src[e] * D_DIM + d]);
    if (d == 0) atomicAdd(&cnt[seg], 1.0f);
}

// ---------------- fused RGCN layer GEMM ------------------------------------
// 256 threads = 8 wave32. Block owns rows m0..m0+79, all 128 cols.
// Wave w owns columns [16w,16w+16) for all 5 row subtiles (5 accumulators),
// reusing one B fragment per 5 WMMAs. B tile arrives via async LDS copy.
__global__ __launch_bounds__(256)
void k_rgcn_gemm(const float* __restrict__ sums, const float* __restrict__ cnt,
                 const float* __restrict__ x,    const __bf16* __restrict__ WT,
                 const float* __restrict__ bias, float* __restrict__ xout) {
    __shared__ __bf16 As [M_TILE][136];  // A tile (80 rows x 128 K), padded
    __shared__ __bf16 BsT[128][136];     // B tile transposed (col-major K), padded

    const int m0   = blockIdx.x * M_TILE;
    const int tid  = threadIdx.x;
    const int lane = tid & 31;
    const int wave = tid >> 5;
    const int half = lane >> 4;          // 0: lanes 0-15, 1: lanes 16-31
    const int mrow = lane & 15;
    const int bcol = wave * 16 + (lane & 15);
    const unsigned lds_b = (unsigned)(size_t)&BsT[0][0];

    v8f acc[5] = {};

    for (int kt = 0; kt < 9; ++kt) {
        // ---- B tile: async global(bf16)->LDS copy, 2048 x 16B units ----
        const __bf16* Bw = WT + (size_t)kt * 128 * 128;
        #pragma unroll
        for (int i = 0; i < 8; ++i) {
            int idx  = tid + i * 256;            // 16-byte unit index
            int col  = idx >> 4;                 // 16 units per 128-elem row
            int unit = idx & 15;
            unsigned ldsoff = lds_b + (unsigned)(col * 272 + unit * 16);
            unsigned long long ga = (unsigned long long)(size_t)(Bw + idx * 8);
            asm volatile("global_load_async_to_lds_b128 %0, %1, off"
                         :: "v"(ldsoff), "v"(ga) : "memory");
        }

        // ---- A tile: 80x128 f32 -> bf16 with 1/cnt scaling; prefetch next ----
        #pragma unroll
        for (int i = 0; i < 10; ++i) {
            int idx = tid + i * 256;             // float4 index, 32 per row
            int row = idx >> 5;
            int c4  = (idx & 31) << 2;
            float4 v;
            if (kt < 8) {
                const float* ap = sums + (size_t)(m0 + row) * 1024 + kt * 128 + c4;
                v = *(const float4*)ap;
                float sc = 1.0f / fmaxf(cnt[(m0 + row) * R_REL + kt], 1.0f);
                v.x *= sc; v.y *= sc; v.z *= sc; v.w *= sc;
                if (kt < 7)       __builtin_prefetch(ap + 128, 0, 3);
                else              __builtin_prefetch(x + (size_t)(m0 + row) * D_DIM + c4, 0, 3);
            } else {
                v = *(const float4*)(x + (size_t)(m0 + row) * D_DIM + c4);
            }
            As[row][c4 + 0] = (__bf16)v.x;  As[row][c4 + 1] = (__bf16)v.y;
            As[row][c4 + 2] = (__bf16)v.z;  As[row][c4 + 3] = (__bf16)v.w;
        }

        asm volatile("s_wait_asynccnt 0x0" ::: "memory");
        __syncthreads();

        // ---- compute: 4 K-steps of 32; 1 B frag reused by 5 A subtiles ----
        #pragma unroll
        for (int kk = 0; kk < 128; kk += 32) {
            v16bf b;
            // B 32x16 bf16: lanes<16 hold K0-15 of their column, lanes>=16 K16-31.
            #pragma unroll
            for (int j = 0; j < 16; ++j) b[j] = BsT[bcol][kk + half * 16 + j];
            #pragma unroll
            for (int s = 0; s < 5; ++s) {
                v16bf a;
                const int ar = s * 16 + mrow;
                // A 16x32 bf16 layout: lanes<16 elems0-15=K0-15;
                // lanes>=16 elems0-15=K16-31.
                #pragma unroll
                for (int j = 0; j < 8; ++j) {
                    a[j]     = As[ar][kk + half * 16 + j];
                    a[j + 8] = As[ar][kk + half * 16 + 8 + j];
                }
                acc[s] = __builtin_amdgcn_wmma_f32_16x16x32_bf16(
                             false, a, false, b, (short)0, acc[s], false, false);
            }
        }
        __syncthreads();
    }

    // ---- epilogue: bias + relu; C/D layout: VGPR q -> row q / 8+q ----
    const int col   = wave * 16 + (lane & 15);
    const int rbase = (lane < 16) ? 0 : 8;
    const float bc  = bias[col];
    #pragma unroll
    for (int s = 0; s < 5; ++s) {
        #pragma unroll
        for (int q = 0; q < 8; ++q) {
            float val = acc[s][q] + bc;
            val = fmaxf(val, 0.0f);
            xout[(size_t)(m0 + s * 16 + rbase + q) * D_DIM + col] = val;
        }
    }
}

// ---------------- global mean pool (sum + count) ----------------
__global__ void k_pool(const int* __restrict__ batch, const float* __restrict__ x,
                       float* __restrict__ gsum, float* __restrict__ gcnt, int N) {
    int t = blockIdx.x * blockDim.x + threadIdx.x;
    if (t >= N * D_DIM) return;
    int n = t >> 7, d = t & 127;
    int b = batch[n];
    atomicAdd(&gsum[(size_t)b * D_DIM + d], x[t]);
    if (d == 0) atomicAdd(&gcnt[b], 1.0f);
}

// ---------------- MLP heads: one block (128 thr) per graph ----------------
__global__ __launch_bounds__(128)
void k_heads(const float* __restrict__ gsum, const float* __restrict__ gcnt,
             const float* __restrict__ rw1, const float* __restrict__ rb1,
             const float* __restrict__ rw2, const float* __restrict__ rb2,
             const float* __restrict__ sw1, const float* __restrict__ sb1,
             const float* __restrict__ sw2, const float* __restrict__ sb2,
             float* __restrict__ out, int B) {
    __shared__ float g[D_DIM];
    __shared__ float red[D_DIM];
    int b = blockIdx.x, d = threadIdx.x;
    g[d] = gsum[(size_t)b * D_DIM + d] / fmaxf(gcnt[b], 1.0f);
    __syncthreads();

    float h = rb1[d];
    for (int k = 0; k < D_DIM; ++k) h += g[k] * rw1[k * D_DIM + d];
    red[d] = fmaxf(h, 0.0f) * rw2[d];
    __syncthreads();
    for (int s = 64; s > 0; s >>= 1) { if (d < s) red[d] += red[d + s]; __syncthreads(); }
    if (d == 0) out[b] = red[0] + rb2[0];
    __syncthreads();

    float h2 = sb1[d];
    for (int k = 0; k < D_DIM; ++k) h2 += g[k] * sw1[k * D_DIM + d];
    red[d] = fmaxf(h2, 0.0f) * sw2[d];
    __syncthreads();
    for (int s = 64; s > 0; s >>= 1) { if (d < s) red[d] += red[d + s]; __syncthreads(); }
    if (d == 0) out[B + b] = red[0] + sb2[0];
}

// ---------------------------------------------------------------------------
extern "C" void kernel_launch(void* const* d_in, const int* in_sizes, int n_in,
                              void* d_out, int out_size, void* d_ws, size_t ws_size,
                              hipStream_t stream) {
    const int N = in_sizes[0];          // 100000
    const int E = in_sizes[2];          // 600000
    const int B = out_size / 2;         // 64

    const int*   node_type = (const int*)d_in[0];
    const int*   src       = (const int*)d_in[1];
    const int*   dst       = ((const int*)d_in[1]) + E;
    const int*   etype     = (const int*)d_in[2];
    const int*   batch     = (const int*)d_in[3];
    const float* node_emb  = (const float*)d_in[4];
    const float* rel_w     = (const float*)d_in[5];   // [L,R,D,D] == [L][1024][128]
    const float* root_w    = (const float*)d_in[6];   // [L,D,D]
    const float* bias      = (const float*)d_in[7];   // [L,D]
    const float* rw1 = (const float*)d_in[8],  *rb1 = (const float*)d_in[9];
    const float* rw2 = (const float*)d_in[10], *rb2 = (const float*)d_in[11];
    const float* sw1 = (const float*)d_in[12], *sb1 = (const float*)d_in[13];
    const float* sw2 = (const float*)d_in[14], *sb2 = (const float*)d_in[15];
    float* out = (float*)d_out;

    // ---- workspace partition (float units) ----
    float* ws   = (float*)d_ws;
    size_t oXA  = 0;
    size_t oXB  = oXA + (size_t)N * D_DIM;
    size_t oSUM = oXB + (size_t)N * D_DIM;
    size_t oCNT = oSUM + (size_t)N * R_REL * D_DIM;   // cnt right after sums
    size_t oGS  = oCNT + (size_t)N * R_REL;
    size_t oGC  = oGS + (size_t)B * D_DIM;
    size_t oWT  = (oGC + (size_t)B + 3) & ~(size_t)3; // 16B-aligned bf16 region
    float*  xA   = ws + oXA;
    float*  xB   = ws + oXB;
    float*  sums = ws + oSUM;
    float*  cnt  = ws + oCNT;
    float*  gsum = ws + oGS;
    float*  gcnt = ws + oGC;
    __bf16* wt   = (__bf16*)(ws + oWT);               // [L][9][128][128] bf16
    (void)ws_size; (void)n_in;

    const int THREADS = 256;
    const int per_layer_w = 9 * 128 * 128;

    // weights -> bf16 transposed chunks (once per launch)
    k_prep_w<<<(L_LAY * per_layer_w + THREADS - 1) / THREADS, THREADS, 0, stream>>>(
        rel_w, root_w, wt);

    // x = node_emb[node_type]
    k_gather_embed<<<(N * D_DIM + THREADS - 1) / THREADS, THREADS, 0, stream>>>(
        node_type, node_emb, xA, N);

    float* xcur = xA;
    float* xnxt = xB;
    const size_t segTotal = (size_t)N * R_REL * D_DIM + (size_t)N * R_REL; // sums+cnt
    for (int l = 0; l < L_LAY; ++l) {
        k_fill_zero<<<8192, THREADS, 0, stream>>>(sums, segTotal);
        k_scatter<<<(E * D_DIM + THREADS - 1) / THREADS, THREADS, 0, stream>>>(
            src, dst, etype, xcur, sums, cnt, E);
        k_rgcn_gemm<<<N / M_TILE, THREADS, 0, stream>>>(
            sums, cnt, xcur,
            wt + (size_t)l * per_layer_w,
            bias + (size_t)l * D_DIM,
            xnxt);
        float* t = xcur; xcur = xnxt; xnxt = t;
    }

    // pool + heads
    k_fill_zero<<<4, THREADS, 0, stream>>>(gsum, (size_t)B * D_DIM + B);
    k_pool<<<(N * D_DIM + THREADS - 1) / THREADS, THREADS, 0, stream>>>(
        batch, xcur, gsum, gcnt, N);
    k_heads<<<B, 128, 0, stream>>>(gsum, gcnt, rw1, rb1, rw2, rb2,
                                   sw1, sb1, sw2, sb2, out, B);
}
